// Attention_50070728737153
// MI455X (gfx1250) — matrix operand order repaired and checked by reading, not compiled, not run on hardware
//
#include <hip/hip_runtime.h>

#define LL 2
#define BB 32
#define SS 2048
#define HH 512
#define DD 1024

typedef __attribute__((ext_vector_type(16))) __bf16          v16bf;
typedef __attribute__((ext_vector_type(16))) unsigned short  v16u;
typedef __attribute__((ext_vector_type(8)))  float           v8f;

// float -> bf16, round-to-nearest-even
__device__ __forceinline__ unsigned short f2bf(float f) {
    unsigned int u = __builtin_bit_cast(unsigned int, f);
    u += 0x7fffu + ((u >> 16) & 1u);
    return (unsigned short)(u >> 16);
}

// cheap tanh: 1 - 2/(exp(2x)+1); uses native v_exp_f32 + v_rcp.
// x->+inf: e=inf -> 1;  x->-inf: e=0 -> -1.
__device__ __forceinline__ float fast_tanh(float x) {
    float e = __expf(2.0f * x);
    return 1.0f - __fdividef(2.0f, e + 1.0f);
}

// async global->LDS 16B per lane, immediate offset applies to BOTH addresses
#define ASYNC_LD16(LA, GA, OFF)                                             \
    asm volatile("global_load_async_to_lds_b128 %0, %1, off offset:" #OFF   \
                 :: "v"(LA), "v"(GA) : "memory")

// ---------------------------------------------------------------------------
// Convert w_e = attn_w[:, D:] (fp32, row-major HxD) into bf16 WMMA-B tiles.
// Layout: [kt][nt][lane][i]  (kt = D/32 k-slices, nt = H/16 n-tiles), so one
// K-slice (all 32 n-tiles for one k-tile) is a contiguous 32KB block that the
// main kernel streams into LDS with async loads.
// Per tile (32x16): lane n (0..15) = column N, lanes 16..31 same N with K+16;
// element i of v16bf = K = kbase + i.
// ---------------------------------------------------------------------------
__global__ void prep_B(const float* __restrict__ attn_w,
                       unsigned short* __restrict__ wsB) {
    int idx   = blockIdx.x * blockDim.x + threadIdx.x;   // 0 .. 524287
    int i     = idx & 15;
    int lane  = (idx >> 4) & 31;
    int nt    = (idx >> 9) & 31;
    int kt    = idx >> 14;
    int n     = lane & 15;
    int kbase = (lane < 16) ? 0 : 16;
    int h     = nt * 16 + n;
    int d     = kt * 32 + kbase + i;
    wsB[idx] = f2bf(attn_w[h * (4 * HH) + DD + d]);
}

// ---------------------------------------------------------------------------
// ehb[b][h] = sum_d hidden[-1][b][d] * attn_w[h][d] + attn_b[h]
// One wave per output, coalesced loads, shfl butterfly reduction (wave32).
// ---------------------------------------------------------------------------
__global__ __launch_bounds__(256) void prep_eh(const float* __restrict__ hidden,
                                               const float* __restrict__ attn_w,
                                               const float* __restrict__ attn_b,
                                               float* __restrict__ ehb) {
    int wid  = threadIdx.x >> 5;
    int lane = threadIdx.x & 31;
    int o    = blockIdx.x * 8 + wid;          // 0 .. 16383
    int b    = o >> 9;
    int hh   = o & (HH - 1);
    const float* hv = hidden + (LL - 1) * BB * DD + b * DD;
    const float* wr = attn_w + hh * (4 * HH);
    float sum = 0.f;
#pragma unroll
    for (int j = 0; j < DD / 32; ++j) {
        int d = lane + j * 32;
        sum += hv[d] * wr[d];
    }
#pragma unroll
    for (int m = 16; m >= 1; m >>= 1) sum += __shfl_xor(sum, m, 32);
    if (lane == 0) ehb[o] = sum + attn_b[hh];
}

// ---------------------------------------------------------------------------
// Main fused kernel: e_e = enc @ w_e^T via bf16 WMMA, fused tanh(+eh+b) and
// dot with v -> scores[b][s].
// Grid: (S/64, B). Block: 256 = 8 waves = mw(0..3: 16 s-rows) x nh(0..1: 256 h).
// B is double-buffered in LDS: one 32KB K-slice per step, moved by
// global_load_async_to_lds_b128 (ASYNCcnt) and consumed via ds_load_b128.
// ---------------------------------------------------------------------------
__global__ __launch_bounds__(256) void attn_scores(
    const float* __restrict__ enc,           // B,S,D fp32
    const unsigned short* __restrict__ wsB,  // tiled bf16 w_e  [kt][nt][lane][i]
    const float* __restrict__ ehb,           // B,H  (e_h + bias)
    const float* __restrict__ vw,            // H
    float* __restrict__ scores)              // B,S
{
    __shared__ unsigned short sB[2][16384];  // 2 x 32KB K-slices of B (64KB)

    int b    = blockIdx.y;
    int s0   = blockIdx.x * 64;
    int wid  = threadIdx.x >> 5;
    int lane = threadIdx.x & 31;
    int mw   = wid & 3;
    int nh   = wid >> 2;
    int m    = lane & 15;
    int hi   = lane >> 4;                    // K sub-range selector for A
    int astart = hi ? 16 : 0;

    const float* arow = enc + ((size_t)b * SS + (s0 + mw * 16 + m)) * DD;

    // each wave moves 8 x (32 lanes x 16B) = 4KB of every 32KB slice
    unsigned chunk = (unsigned)((wid * 8) * 32 + lane) * 16;      // byte off in slice
    const char* gsrc = (const char*)wsB;
    unsigned lds0 = (unsigned)(uintptr_t)(&sB[0][0]);
    unsigned lds1 = (unsigned)(uintptr_t)(&sB[1][0]);

    // prologue: issue slice 0 -> buffer 0
    {
        unsigned long long ga = (unsigned long long)(uintptr_t)(gsrc + chunk);
        unsigned la = lds0 + chunk;
        ASYNC_LD16(la, ga, 0);    ASYNC_LD16(la, ga, 512);
        ASYNC_LD16(la, ga, 1024); ASYNC_LD16(la, ga, 1536);
        ASYNC_LD16(la, ga, 2048); ASYNC_LD16(la, ga, 2560);
        ASYNC_LD16(la, ga, 3072); ASYNC_LD16(la, ga, 3584);
    }

    v8f acc[16];
    const v8f vzero = {0.f, 0.f, 0.f, 0.f, 0.f, 0.f, 0.f, 0.f};
#pragma unroll
    for (int nt = 0; nt < 16; ++nt) acc[nt] = vzero;

    for (int kt = 0; kt < DD / 32; ++kt) {
        // issue next slice into the other buffer, then wait for current slice.
        if (kt + 1 < DD / 32) {
            unsigned long long ga = (unsigned long long)(uintptr_t)
                (gsrc + (size_t)(kt + 1) * 32768 + chunk);
            unsigned la = (((kt + 1) & 1) ? lds1 : lds0) + chunk;
            ASYNC_LD16(la, ga, 0);    ASYNC_LD16(la, ga, 512);
            ASYNC_LD16(la, ga, 1024); ASYNC_LD16(la, ga, 1536);
            ASYNC_LD16(la, ga, 2048); ASYNC_LD16(la, ga, 2560);
            ASYNC_LD16(la, ga, 3072); ASYNC_LD16(la, ga, 3584);
            // async loads complete in order: cnt<=8 => current slice landed
            asm volatile("s_wait_asynccnt 0x8" ::: "memory");
        } else {
            asm volatile("s_wait_asynccnt 0x0" ::: "memory");
        }
        __syncthreads();   // all waves' portions of current slice visible

        // ---- load + convert A tile (16x32 bf16, element i = k 16*hi + i) ----
        int k0 = kt * 32;
        const float4* p0 = (const float4*)(arow + k0 + astart);
        const float4* p1 = (const float4*)(arow + k0 + astart + 8);
        float4 f0 = p0[0], f1 = p0[1];
        float4 f2 = p1[0], f3 = p1[1];
        __builtin_prefetch(arow + k0 + 32 + astart, 0, 0);
        v16u au;
        au[0]  = f2bf(f0.x); au[1]  = f2bf(f0.y); au[2]  = f2bf(f0.z); au[3]  = f2bf(f0.w);
        au[4]  = f2bf(f1.x); au[5]  = f2bf(f1.y); au[6]  = f2bf(f1.z); au[7]  = f2bf(f1.w);
        au[8]  = f2bf(f2.x); au[9]  = f2bf(f2.y); au[10] = f2bf(f2.z); au[11] = f2bf(f2.w);
        au[12] = f2bf(f3.x); au[13] = f2bf(f3.y); au[14] = f2bf(f3.z); au[15] = f2bf(f3.w);
        v16bf a = __builtin_bit_cast(v16bf, au);

        const unsigned short* bk = &sB[kt & 1][0] + (nh * 16) * 512 + lane * 16;
#pragma unroll
        for (int nt = 0; nt < 16; ++nt) {
            v16u bu = *(const v16u*)(bk + nt * 512);      // 2x ds_load_b128
            v16bf bt = __builtin_bit_cast(v16bf, bu);
            acc[nt] = __builtin_amdgcn_wmma_f32_16x16x32_bf16(
                false, a, false, bt, (short)0, acc[nt], false, false);
        }
        __syncthreads();   // slice consumed; safe to overwrite next round
    }

    // ---- epilogue: tanh(acc + ehb) dot v, reduce across H ----
    float part[8];
#pragma unroll
    for (int r = 0; r < 8; ++r) part[r] = 0.f;
#pragma unroll
    for (int nt = 0; nt < 16; ++nt) {
        int h    = (nh * 16 + nt) * 16 + (lane & 15);
        float eh = ehb[b * HH + h];
        float vv = vw[h];
#pragma unroll
        for (int r = 0; r < 8; ++r) {
            float e = fast_tanh(acc[nt][r] + eh);
            part[r] += e * vv;
        }
    }
    // sB no longer needed (final barrier passed) -> reuse as reduction scratch
    float* sred = (float*)&sB[0][0];   // [2][64]
#pragma unroll
    for (int r = 0; r < 8; ++r) {
        float p = part[r];
#pragma unroll
        for (int msk = 8; msk >= 1; msk >>= 1) p += __shfl_xor(p, msk, 32);
        if ((lane & 15) == 0) sred[nh * 64 + mw * 16 + r + hi * 8] = p;
    }
    __syncthreads();
    int t = threadIdx.x;
    if (t < 64) scores[b * SS + s0 + t] = sred[t] + sred[64 + t];
}

// ---------------------------------------------------------------------------
// Softmax over S per batch; writes attention weights into the output buffer.
// ---------------------------------------------------------------------------
__global__ __launch_bounds__(256) void softmax_k(const float* __restrict__ scores,
                                                 float* __restrict__ wout) {
    __shared__ float red[256];
    int b = blockIdx.x;
    int t = threadIdx.x;
    float v[8];
    float m = -1e30f;
#pragma unroll
    for (int j = 0; j < 8; ++j) {
        v[j] = scores[b * SS + t + j * 256];
        m = fmaxf(m, v[j]);
    }
    red[t] = m; __syncthreads();
    for (int off = 128; off > 0; off >>= 1) {
        if (t < off) red[t] = fmaxf(red[t], red[t + off]);
        __syncthreads();
    }
    m = red[0]; __syncthreads();
    float sum = 0.f;
#pragma unroll
    for (int j = 0; j < 8; ++j) { v[j] = __expf(v[j] - m); sum += v[j]; }
    red[t] = sum; __syncthreads();
    for (int off = 128; off > 0; off >>= 1) {
        if (t < off) red[t] += red[t + off];
        __syncthreads();
    }
    float inv = 1.f / red[0];
#pragma unroll
    for (int j = 0; j < 8; ++j) wout[b * SS + t + j * 256] = v[j] * inv;
}

// ---------------------------------------------------------------------------
// context partials: each block = (s-chunk of 128, batch b), 256 threads x
// float4 covers D=1024. Coalesced streaming of enc (bandwidth bound).
// ---------------------------------------------------------------------------
__global__ __launch_bounds__(256) void ctx_part(const float* __restrict__ enc,
                                                const float* __restrict__ w,
                                                float* __restrict__ part) {
    int sc = blockIdx.x, b = blockIdx.y, t = threadIdx.x;
    int d = t * 4;
    float4 acc = {0.f, 0.f, 0.f, 0.f};
    const float* ebase = enc + ((size_t)b * SS + sc * 128) * DD + d;
    const float* wb    = w + b * SS + sc * 128;
    for (int s = 0; s < 128; ++s) {
        float ws  = wb[s];
        float4 ev = *(const float4*)(ebase + (size_t)s * DD);
        acc.x += ws * ev.x; acc.y += ws * ev.y;
        acc.z += ws * ev.z; acc.w += ws * ev.w;
    }
    *(float4*)(part + ((size_t)(b * 16 + sc) * DD) + d) = acc;
}

__global__ void ctx_reduce(const float* __restrict__ part,
                           float* __restrict__ out) {
    int idx = blockIdx.x * blockDim.x + threadIdx.x;  // 0 .. 32767
    int b = idx >> 10, d = idx & (DD - 1);
    float s = 0.f;
#pragma unroll
    for (int sc = 0; sc < 16; ++sc) s += part[(size_t)(b * 16 + sc) * DD + d];
    out[idx] = s;
}

// ---------------------------------------------------------------------------
extern "C" void kernel_launch(void* const* d_in, const int* in_sizes, int n_in,
                              void* d_out, int out_size, void* d_ws, size_t ws_size,
                              hipStream_t stream) {
    (void)in_sizes; (void)n_in; (void)out_size; (void)ws_size;
    const float* hidden = (const float*)d_in[0];
    const float* enc    = (const float*)d_in[1];
    const float* attn_w = (const float*)d_in[2];
    const float* attn_b = (const float*)d_in[3];
    const float* v_w    = (const float*)d_in[4];

    float* out_ctx = (float*)d_out;            // (32,1,1024)
    float* out_w   = (float*)d_out + BB * DD;  // (32,1,2048)

    char* ws = (char*)d_ws;
    unsigned short* wsB = (unsigned short*)ws;                                 // 1 MB
    float* ehb    = (float*)(ws + (1u << 20));                                 // 64 KB
    float* scores = (float*)(ws + (1u << 20) + (64u << 10));                   // 256 KB
    float* part   = (float*)(ws + (1u << 20) + (64u << 10) + (256u << 10));    // 2 MB

    prep_B     <<<2048, 256, 0, stream>>>(attn_w, wsB);
    prep_eh    <<<2048, 256, 0, stream>>>(hidden, attn_w, attn_b, ehb);
    attn_scores<<<dim3(SS / 64, BB), 256, 0, stream>>>(enc, wsB, ehb, v_w, scores);
    softmax_k  <<<BB, 256, 0, stream>>>(scores, out_w);
    ctx_part   <<<dim3(16, BB), 256, 0, stream>>>(enc, out_w, part);
    ctx_reduce <<<128, 256, 0, stream>>>(part, out_ctx);
}
